// Quantizer_80942953660682
// MI455X (gfx1250) — matrix operand order repaired and checked by reading, compiled, not run on hardware
//
#include <hip/hip_runtime.h>
#include <hip/hip_bf16.h>

// CDNA5 / gfx1250, wave32.
typedef __attribute__((ext_vector_type(16))) __bf16 v16bf;
typedef __attribute__((ext_vector_type(8)))  __bf16 v8bf;
typedef __attribute__((ext_vector_type(8)))  float  v8f;

#define B_  16
#define D_  256
#define HW_ 4096                    // 64*64
#define K_  512
#define NTOT (B_ * HW_)             // 65536 positions
#define XELEMS ((size_t)NTOT * D_)  // 16777216 floats of x output

#define WAVES_PER_BLOCK 8
#define ROWSTRIDE 264               // 256 bf16 + 8 pad (16B-aligned rows, bank spread)

#if defined(__AMDGCN__) && __has_builtin(__builtin_amdgcn_global_load_async_to_lds_b128)
#define USE_ASYNC_LDS 1
#else
#define USE_ASYNC_LDS 0
#endif

// ---------------------------------------------------------------------------
// Prep: split codebook fp32 -> bf16 hi/lo pair (c = hi + lo), and |c|^2 per row.
// ---------------------------------------------------------------------------
__global__ __launch_bounds__(256) void vq_prep(const float* __restrict__ cb,
                                               __bf16* __restrict__ chi,
                                               __bf16* __restrict__ clo,
                                               float* __restrict__ csq) {
  __shared__ float red[8];
  const int k = blockIdx.x;
  const int d = threadIdx.x;
  const float c = cb[(size_t)k * D_ + d];
  const __bf16 h = (__bf16)c;
  chi[(size_t)k * D_ + d] = h;
  clo[(size_t)k * D_ + d] = (__bf16)(c - (float)h);
  float s = c * c;
#pragma unroll
  for (int m = 16; m >= 1; m >>= 1) s += __shfl_xor(s, m, 32);
  if ((d & 31) == 0) red[d >> 5] = s;
  __syncthreads();
  if (d == 0) {
    float t = 0.f;
#pragma unroll
    for (int w = 0; w < 8; ++w) t += red[w];
    csq[k] = t;
  }
}

// Per-thread slice of a B-tile stage: 16KB tile, 256 threads x 4 x 16B.
struct StageAddr {
  size_t g;  // element offset into chi/clo
  int    s;  // element offset into LDS buffer
};
__device__ __forceinline__ StageAddr stage_addr(int k0, int tid) {
  StageAddr a;
  const int row = tid >> 4;
  const int c16 = (tid & 15) << 4;
  a.g = ((size_t)(k0 + row) << 8) + (size_t)c16;
  a.s = row * ROWSTRIDE + c16;
  return a;
}

#if USE_ASYNC_LDS
// Memory -> LDS DMA, no VGPR round trip; tracked with ASYNCcnt.
typedef int v4i_vs __attribute__((vector_size(16)));
typedef __attribute__((address_space(1))) v4i_vs* gv4i_p;
typedef __attribute__((address_space(3))) v4i_vs* lv4i_p;

__device__ __forceinline__ void async_cp16(const void* g, void* l) {
  __builtin_amdgcn_global_load_async_to_lds_b128(
      (gv4i_p)(uintptr_t)g, (lv4i_p)(uintptr_t)l, 0, 0);
}
__device__ __forceinline__ void wait_async0() {
#if __has_builtin(__builtin_amdgcn_s_wait_asynccnt)
  __builtin_amdgcn_s_wait_asynccnt(0);
#else
  asm volatile("s_wait_asynccnt 0" ::: "memory");
#endif
}
__device__ __forceinline__ void stage_tile_async(const __bf16* chi, const __bf16* clo,
                                                 __bf16* sh, __bf16* sl,
                                                 int k0, int tid) {
  const StageAddr a = stage_addr(k0, tid);
  async_cp16(chi + a.g,     sh + a.s);
  async_cp16(chi + a.g + 8, sh + a.s + 8);
  async_cp16(clo + a.g,     sl + a.s);
  async_cp16(clo + a.g + 8, sl + a.s + 8);
}
#else
// Fallback: load into registers at the top of the iteration, store to LDS
// after the WMMA chain so the global-load latency is hidden by compute.
struct StageRegs { uint4 h0, h1, l0, l1; };
__device__ __forceinline__ StageRegs stage_load(const __bf16* chi, const __bf16* clo,
                                                int k0, int tid) {
  const StageAddr a = stage_addr(k0, tid);
  StageRegs r;
  r.h0 = *(const uint4*)(chi + a.g);
  r.h1 = *(const uint4*)(chi + a.g + 8);
  r.l0 = *(const uint4*)(clo + a.g);
  r.l1 = *(const uint4*)(clo + a.g + 8);
  return r;
}
__device__ __forceinline__ void stage_store(const StageRegs& r, __bf16* sh, __bf16* sl,
                                            int tid) {
  const StageAddr a = stage_addr(0, tid);
  *(uint4*)(sh + a.s)     = r.h0;
  *(uint4*)(sh + a.s + 8) = r.h1;
  *(uint4*)(sl + a.s)     = r.l0;
  *(uint4*)(sl + a.s + 8) = r.l1;
}
#endif

__device__ __forceinline__ v16bf lds_b_operand(const __bf16* s, int off) {
  const v8bf x0 = *(const v8bf*)(s + off);
  const v8bf x1 = *(const v8bf*)(s + off + 16);
  return __builtin_shufflevector(x0, x1, 0, 1, 2, 3, 4, 5, 6, 7,
                                 8, 9, 10, 11, 12, 13, 14, 15);
}

// ---------------------------------------------------------------------------
// Main: 8 waves/block, each wave owns a 16-position tile (M=128 per block).
// B tiles staged once per block into LDS (double-buffered, async DMA) -> 8x
// less L2 traffic. Cross term via split-bf16 WMMA (hi*hi + lo*hi + hi*lo).
// argmin_k(|c_k|^2 - 2*z.c_k), then exact fp32 codebook gather.
// ---------------------------------------------------------------------------
__global__ __launch_bounds__(256) void vq_main(const float* __restrict__ z,
                                               const float* __restrict__ cb,
                                               const __bf16* __restrict__ chi,
                                               const __bf16* __restrict__ clo,
                                               const float* __restrict__ csq,
                                               float* __restrict__ xout,
                                               int* __restrict__ iout) {
  __shared__ __bf16 sb[2][2][16 * ROWSTRIDE];  // [buf][hi/lo][tile], ~33 KB

  const int tid     = threadIdx.x;
  const int wave    = tid >> 5;
  const int lane    = tid & 31;
  const int p0      = (blockIdx.x * WAVES_PER_BLOCK + wave) * 16;
  const int b       = p0 >> 12;          // / HW_
  const int n0      = p0 & (HW_ - 1);
  const int rowm    = lane & 15;
  const int hi_half = lane >> 4;

  // ---- Load A tile (16 rows x 256 feats) into WMMA 16-bit A layout --------
  // A 16x32 bf16: lanes 0-15 hold row M=lane, K = {0..7, 16..23};
  //               lanes 16-31 hold row M=lane-16, K = {8..15, 24..31}.
  // z is [B, D, HW]: element (b, d, n) at b*D*HW + d*HW + n.
  const float* zb = z + ((size_t)b << 20) + (size_t)n0 + rowm;
  v16bf a_hi[8], a_lo[8];
  const int dsh = hi_half * 8;
#pragma unroll
  for (int kk = 0; kk < 8; ++kk) {
#pragma unroll
    for (int j = 0; j < 8; ++j) {
      const float f0 = zb[(size_t)(kk * 32 + dsh + j) << 12];
      const __bf16 h0 = (__bf16)f0;
      a_hi[kk][j] = h0;
      a_lo[kk][j] = (__bf16)(f0 - (float)h0);
      const float f1 = zb[(size_t)(kk * 32 + 16 + dsh + j) << 12];
      const __bf16 h1 = (__bf16)f1;
      a_hi[kk][8 + j] = h1;
      a_lo[kk][8 + j] = (__bf16)(f1 - (float)h1);
    }
  }

  float best[8];
  int   bestk[8];
#pragma unroll
  for (int r = 0; r < 8; ++r) { best[r] = 3.402823466e38f; bestk[r] = 0; }

  // ---- Sweep 32 code tiles of 16 codes, double-buffered through LDS -------
#if USE_ASYNC_LDS
  stage_tile_async(chi, clo, sb[0][0], sb[0][1], 0, tid);
  wait_async0();
  __syncthreads();
#else
  {
    const StageRegs r0 = stage_load(chi, clo, 0, tid);
    stage_store(r0, sb[0][0], sb[0][1], tid);
  }
  __syncthreads();
#endif

  for (int t = 0; t < 32; ++t) {
    const int cur = t & 1;

#if USE_ASYNC_LDS
    if (t < 31)
      stage_tile_async(chi, clo, sb[cur ^ 1][0], sb[cur ^ 1][1], (t + 1) * 16, tid);
#else
    StageRegs r;
    if (t < 31) r = stage_load(chi, clo, (t + 1) * 16, tid);
#endif

    const __bf16* sh = sb[cur][0];
    const __bf16* sl = sb[cur][1];
    const int k0 = t * 16;

    // B 32x16 bf16 layout: lanes 0-15 col N=lane hold K={0..7, 16..23},
    // lanes 16-31 col N=lane-16 hold K={8..15, 24..31} (matches A).
    const int bbase = rowm * ROWSTRIDE + hi_half * 8;

    v8f acc = {0.f, 0.f, 0.f, 0.f, 0.f, 0.f, 0.f, 0.f};
#pragma unroll
    for (int kk = 0; kk < 8; ++kk) {
      const v16bf bh = lds_b_operand(sh, bbase + kk * 32);
      const v16bf bl = lds_b_operand(sl, bbase + kk * 32);
      acc = __builtin_amdgcn_wmma_f32_16x16x32_bf16(false, a_hi[kk], false, bh,
                                                    (short)0, acc, false, false);
      acc = __builtin_amdgcn_wmma_f32_16x16x32_bf16(false, a_lo[kk], false, bh,
                                                    (short)0, acc, false, false);
      acc = __builtin_amdgcn_wmma_f32_16x16x32_bf16(false, a_hi[kk], false, bl,
                                                    (short)0, acc, false, false);
    }

    // C layout: lane n (0-15) VGPR r -> (M=r, N=n); lane n+16 -> (M=8+r, N=n)
    const float cs    = csq[k0 + rowm];
    const int   kcand = k0 + rowm;
#pragma unroll
    for (int r8 = 0; r8 < 8; ++r8) {
      const float s = cs - 2.0f * acc[r8];
      if (s < best[r8]) { best[r8] = s; bestk[r8] = kcand; }  // strict <: first occurrence
    }

#if USE_ASYNC_LDS
    if (t < 31) wait_async0();   // next tile fully landed in LDS
#else
    if (t < 31) stage_store(r, sb[cur ^ 1][0], sb[cur ^ 1][1], tid);
#endif
    __syncthreads();
  }

  // ---- Cross-lane argmin within each 16-lane group (ties -> smaller k) ----
#pragma unroll
  for (int r = 0; r < 8; ++r) {
    float dv = best[r];
    int   iv = bestk[r];
#pragma unroll
    for (int m = 1; m < 16; m <<= 1) {
      const float od = __shfl_xor(dv, m, 32);
      const int   oi = __shfl_xor(iv, m, 32);
      if (od < dv || (od == dv && oi < iv)) { dv = od; iv = oi; }
    }
    best[r] = dv;
    bestk[r] = iv;
  }

  // ---- Write indices + gather winning codebook rows (exact fp32) ----------
#pragma unroll
  for (int m = 0; m < 16; ++m) {
    const int idx_m = __shfl(bestk[m & 7], (m < 8) ? 0 : 16, 32);
    if (lane == m) iout[p0 + m] = idx_m;
    const float4* src = (const float4*)(cb + ((size_t)idx_m << 8));
    float4*       dst = (float4*)(xout + ((size_t)(p0 + m) << 8));
    dst[lane]      = src[lane];       // 32 lanes x 16B = first 128 floats
    dst[lane + 32] = src[lane + 32];  // next 128 floats
  }
}

// ---------------------------------------------------------------------------
extern "C" void kernel_launch(void* const* d_in, const int* in_sizes, int n_in,
                              void* d_out, int out_size, void* d_ws, size_t ws_size,
                              hipStream_t stream) {
  (void)in_sizes; (void)n_in; (void)out_size; (void)ws_size;
  const float* z  = (const float*)d_in[0];   // [16, 256, 64, 64] fp32
  const float* cb = (const float*)d_in[1];   // [512, 256] fp32

  // workspace: chi (256KB) | clo (256KB) | csq (2KB)
  __bf16* chi = (__bf16*)d_ws;
  __bf16* clo = chi + (size_t)K_ * D_;
  float*  csq = (float*)(clo + (size_t)K_ * D_);

  float* xout = (float*)d_out;               // [16, 4096, 256] fp32
  int*   iout = (int*)(xout + XELEMS);       // [16, 4096] int32

  vq_prep<<<K_, 256, 0, stream>>>(cb, chi, clo, csq);
  vq_main<<<NTOT / (16 * WAVES_PER_BLOCK), 256, 0, stream>>>(
      z, cb, chi, clo, csq, xout, iout);
}
